// CrossAttention_17987323035795
// MI455X (gfx1250) — compile-verified, matrix operand order checked by reading, not run on hardware
//
#include <hip/hip_runtime.h>

// ---------------- problem constants ----------------
#define BATCH 2
#define NQ    2048
#define NKV   2048
#define QDIM  1024
#define KVDIM 1024
#define H     16
#define HD    64
#define INNER (H * HD)        // 1024
#define MQ    (BATCH * NQ)    // 4096
#define MKV   (BATCH * NKV)   // 4096

typedef _Float16 h16;
typedef h16   v16h __attribute__((ext_vector_type(16)));
typedef h16   v8h  __attribute__((ext_vector_type(8)));
typedef float v8f  __attribute__((ext_vector_type(8)));

// ---- TDM availability (this toolchain: 6-arg clang-23 form) ----
#if defined(__has_builtin)
#  if __has_builtin(__builtin_amdgcn_tensor_load_to_lds) && \
      __has_builtin(__builtin_amdgcn_s_wait_tensorcnt)
#    define HAVE_TDM 1
#  endif
#endif
#ifndef HAVE_TDM
#  define HAVE_TDM 0
#endif

// ---- WMMA fragment from K-fast memory (A rows or B^T rows) ----
// element i -> K = (i&7) + (i>=8?16:0) + 8*hi : two contiguous 16B runs
// base must be 16B-aligned; lowers to two global_load_b128 / ds_load_b128.
__device__ __forceinline__ v16h frag_kfast(const h16* __restrict__ base, int hi) {
  const v8h lo = *(const v8h*)(base + hi * 8);
  const v8h hi8 = *(const v8h*)(base + hi * 8 + 16);
  return __builtin_shufflevector(lo, hi8, 0, 1, 2, 3, 4, 5, 6, 7,
                                 8, 9, 10, 11, 12, 13, 14, 15);
}

// ---------------- f32 -> f16 convert ----------------
__global__ void f32_to_f16(const float* __restrict__ in, h16* __restrict__ out, int n) {
  int i = blockIdx.x * blockDim.x + threadIdx.x;
  if (i < n) out[i] = (h16)in[i];
}

// ---------------- f32 -> f16 transpose (weights, done once) ----------------
__global__ __launch_bounds__(256)
void transpose_f32_to_f16(const float* __restrict__ in, h16* __restrict__ out,
                          int R, int C) {
  __shared__ h16 tile[32][33];
  int c0 = blockIdx.x * 32, r0 = blockIdx.y * 32;
  int tx = threadIdx.x;
#pragma unroll
  for (int i = threadIdx.y; i < 32; i += 8)
    tile[i][tx] = (h16)in[(size_t)(r0 + i) * C + c0 + tx];
  __syncthreads();
#pragma unroll
  for (int i = threadIdx.y; i < 32; i += 8)
    out[(size_t)(c0 + i) * R + r0 + tx] = tile[tx][i];
}

// ---------------- f16 GEMM: A[M][K] x BT[N][K]^T, f16 out ----------------
__global__ __launch_bounds__(256)
void gemm_tt_h16(const h16* __restrict__ A, const h16* __restrict__ BT,
                 h16* __restrict__ C, int N, int K) {
  int lane = threadIdx.x & 31, wave = threadIdx.x >> 5;
  int hi = lane >> 4, l15 = lane & 15;
  int wm = wave & 3, wn = wave >> 2;
  int tm = blockIdx.y * 128 + wm * 32;
  int tn = blockIdx.x * 64 + wn * 32;
  const h16* a0 = A + (size_t)(tm + l15) * K;
  const h16* a1 = a0 + (size_t)16 * K;
  const h16* b0 = BT + (size_t)(tn + l15) * K;
  const h16* b1 = b0 + (size_t)16 * K;
  v8f acc00 = {}, acc01 = {}, acc10 = {}, acc11 = {};
  for (int k0 = 0; k0 < K; k0 += 32) {
    __builtin_prefetch((const void*)(a0 + k0 + 128), 0, 1);  // global_prefetch_b8
    __builtin_prefetch((const void*)(b0 + k0 + 128), 0, 1);
    v16h af0 = frag_kfast(a0 + k0, hi);
    v16h af1 = frag_kfast(a1 + k0, hi);
    v16h bf0 = frag_kfast(b0 + k0, hi);
    v16h bf1 = frag_kfast(b1 + k0, hi);
    acc00 = __builtin_amdgcn_wmma_f32_16x16x32_f16(false, af0, false, bf0, (short)0, acc00, false, false);
    acc01 = __builtin_amdgcn_wmma_f32_16x16x32_f16(false, af0, false, bf1, (short)0, acc01, false, false);
    acc10 = __builtin_amdgcn_wmma_f32_16x16x32_f16(false, af1, false, bf0, (short)0, acc10, false, false);
    acc11 = __builtin_amdgcn_wmma_f32_16x16x32_f16(false, af1, false, bf1, (short)0, acc11, false, false);
  }
  v8f* accs[2][2] = {{&acc00, &acc01}, {&acc10, &acc11}};
#pragma unroll
  for (int mi = 0; mi < 2; ++mi)
#pragma unroll
    for (int ni = 0; ni < 2; ++ni) {
      h16* outp = C + (size_t)(tm + mi * 16 + hi * 8) * N + tn + ni * 16 + l15;
#pragma unroll
      for (int r = 0; r < 8; ++r) outp[(size_t)r * N] = (h16)((*accs[mi][ni])[r]);
    }
}

// ---------------- f16 GEMM, f32 out + bias (output projection) ----------------
__global__ __launch_bounds__(256)
void gemm_tt_f32_bias(const h16* __restrict__ A, const h16* __restrict__ BT,
                      const float* __restrict__ bias, float* __restrict__ C,
                      int N, int K) {
  int lane = threadIdx.x & 31, wave = threadIdx.x >> 5;
  int hi = lane >> 4, l15 = lane & 15;
  int wm = wave & 3, wn = wave >> 2;
  int tm = blockIdx.y * 128 + wm * 32;
  int tn = blockIdx.x * 64 + wn * 32;
  const h16* a0 = A + (size_t)(tm + l15) * K;
  const h16* a1 = a0 + (size_t)16 * K;
  const h16* b0 = BT + (size_t)(tn + l15) * K;
  const h16* b1 = b0 + (size_t)16 * K;
  v8f acc00 = {}, acc01 = {}, acc10 = {}, acc11 = {};
  for (int k0 = 0; k0 < K; k0 += 32) {
    __builtin_prefetch((const void*)(a0 + k0 + 128), 0, 1);
    __builtin_prefetch((const void*)(b0 + k0 + 128), 0, 1);
    v16h af0 = frag_kfast(a0 + k0, hi);
    v16h af1 = frag_kfast(a1 + k0, hi);
    v16h bf0 = frag_kfast(b0 + k0, hi);
    v16h bf1 = frag_kfast(b1 + k0, hi);
    acc00 = __builtin_amdgcn_wmma_f32_16x16x32_f16(false, af0, false, bf0, (short)0, acc00, false, false);
    acc01 = __builtin_amdgcn_wmma_f32_16x16x32_f16(false, af0, false, bf1, (short)0, acc01, false, false);
    acc10 = __builtin_amdgcn_wmma_f32_16x16x32_f16(false, af1, false, bf0, (short)0, acc10, false, false);
    acc11 = __builtin_amdgcn_wmma_f32_16x16x32_f16(false, af1, false, bf1, (short)0, acc11, false, false);
  }
  v8f* accs[2][2] = {{&acc00, &acc01}, {&acc10, &acc11}};
#pragma unroll
  for (int mi = 0; mi < 2; ++mi)
#pragma unroll
    for (int ni = 0; ni < 2; ++ni) {
      float bv = bias[tn + ni * 16 + l15];
      float* outp = C + (size_t)(tm + mi * 16 + hi * 8) * N + tn + ni * 16 + l15;
#pragma unroll
      for (int r = 0; r < 8; ++r) outp[(size_t)r * N] = (*accs[mi][ni])[r] + bv;
    }
}

// ---------------- TDM descriptor: 32(kv) x 64(d) f16 tile -> LDS ----------------
// tensor row pitch = 2*INNER elements; LDS padded 16B per 128B row (pad codes:
// interval 4 -> 32 dwords = one row, amount 3 -> 4 dwords = 16B = VPAD).
#define VPAD 8
#if HAVE_TDM
typedef unsigned u32x4 __attribute__((ext_vector_type(4)));
typedef int      i32x8 __attribute__((ext_vector_type(8)));
typedef int      i32x4 __attribute__((ext_vector_type(4)));

__device__ __forceinline__ void tdm_load_v_tile(unsigned lds_off, const h16* g) {
  unsigned long long ga = (unsigned long long)(size_t)g;
  u32x4 g0 = { 1u,                                   // count=1 (valid descriptor)
               lds_off,                               // lds_addr (bytes)
               (unsigned)ga,                          // global_addr[31:0]
               ((unsigned)(ga >> 32) & 0x01FFFFFFu)   // global_addr[56:32]
                 | 0x80000000u };                     // type=2 ("image")
  i32x8 g1 = { (1 << 16) | (1 << 20) | (4 << 22) | (3 << 25), // data_size=2B, pad en, interval=32dw, amount=4dw
               (int)(64u << 16),   // tensor_dim0 = 64 (d elements per row)
               (int)(32u << 16),   // tensor_dim0 hi=0 | tensor_dim1 = 32 rows
               (int)(64u << 16),   // tensor_dim1 hi=0 | tile_dim0 = 64
               32,                 // tile_dim1 = 32 | tile_dim2 = 0
               2 * INNER,          // tensor_dim0_stride = 2048 elements
               0, 0 };
  i32x4 z4 = { 0, 0, 0, 0 };
  i32x8 z8 = { 0, 0, 0, 0, 0, 0, 0, 0 };
  // clang-23 / therock form: (g0, g1, g2, g3, extra, cpol)
  __builtin_amdgcn_tensor_load_to_lds(g0, g1, z4, z4, z8, 0);
}
#endif

// ---------------- flash attention, transposed-score formulation ----------------
// grid: (NQ/16, H, BATCH), block 32 (one wave; 16 query rows)
__global__ __launch_bounds__(32)
void attn(const h16* __restrict__ Q, const h16* __restrict__ KV, h16* __restrict__ CTX) {
  int lane = threadIdx.x;
  int hi = lane >> 4, l15 = lane & 15;
  int q0 = blockIdx.x * 16;
  int hh = blockIdx.y;
  int bb = blockIdx.z;

  const h16* Qb = Q  + ((size_t)bb * NQ)  * INNER       + hh * HD;  // [q][d]
  const h16* Kb = KV + ((size_t)bb * NKV) * (2 * INNER) + hh * HD;  // [kv][d]
  const h16* Vb = Kb + INNER;

  __shared__ h16 Vs[2][32 * (HD + VPAD)];  // double-buffered padded V tiles

  // Loop-invariant B-fragments of Q^T (lane -> col q, K = d contiguous)
  v16h bq0 = frag_kfast(Qb + (size_t)(q0 + l15) * INNER, hi);
  v16h bq1 = frag_kfast(Qb + (size_t)(q0 + l15) * INNER + 32, hi);

  float m = -INFINITY, lsum = 0.f;
  v8f o[4] = {};
  const float sc  = 0.125f;                  // 1/sqrt(HD)
  const float L2E = 1.44269504088896340736f; // log2(e)

#if HAVE_TDM
  tdm_load_v_tile((unsigned)(size_t)&Vs[0][0], Vb);  // prologue: tile 0 in flight
#endif

  for (int kv0 = 0; kv0 < NKV; kv0 += 32) {
    int cur = (kv0 >> 5) & 1;
#if HAVE_TDM
    if (kv0 + 32 < NKV)  // pipeline: DMA next tile while computing this one
      tdm_load_v_tile((unsigned)(size_t)&Vs[cur ^ 1][0],
                      Vb + (size_t)(kv0 + 32) * (2 * INNER));
#else
    {
      const uint4* s4 = (const uint4*)(Vb + (size_t)(kv0 + lane) * (2 * INNER));
      uint4* dst = (uint4*)(&Vs[cur][lane * (HD + VPAD)]);
      dst[0] = s4[0]; dst[1] = s4[1]; dst[2] = s4[2]; dst[3] = s4[3];
    }
#endif
    __builtin_prefetch((const void*)(Kb + (size_t)(kv0 + 32 + lane) * (2 * INNER)), 0, 1);

    // S^T = K * Q^T : two 16(kv)x16(q) tiles, K-dim d=64 split in two
    v8f st[2];
#pragma unroll
    for (int kvT = 0; kvT < 2; ++kvT) {
      const h16* krow = Kb + (size_t)(kv0 + kvT * 16 + l15) * (2 * INNER);
      v8f c = {};
      c = __builtin_amdgcn_wmma_f32_16x16x32_f16(false, frag_kfast(krow, hi),      false, bq0, (short)0, c, false, false);
      c = __builtin_amdgcn_wmma_f32_16x16x32_f16(false, frag_kfast(krow + 32, hi), false, bq1, (short)0, c, false, false);
      st[kvT] = c;
    }

    // streaming softmax: columns of S^T are lane-local + one shfl_xor(16)
    float tmax = -INFINITY;
#pragma unroll
    for (int kvT = 0; kvT < 2; ++kvT)
#pragma unroll
      for (int r = 0; r < 8; ++r) {
        float e = st[kvT][r] * sc;
        st[kvT][r] = e;
        tmax = fmaxf(tmax, e);
      }
    tmax = fmaxf(tmax, __shfl_xor(tmax, 16, 32));
    float mn   = fmaxf(m, tmax);
    float corr = __builtin_exp2f((m - mn) * L2E);
    float rs = 0.f;
    v16h pb;  // P^T: element i -> kv=(i&7)+(i>=8?16:0)+8*hi == C-layout of S^T
#pragma unroll
    for (int kvT = 0; kvT < 2; ++kvT)
#pragma unroll
      for (int r = 0; r < 8; ++r) {
        float p = __builtin_exp2f((st[kvT][r] - mn) * L2E);
        rs += p;
        pb[kvT * 8 + r] = (h16)p;
      }
    rs += __shfl_xor(rs, 16, 32);
    lsum = lsum * corr + rs;
    m = mn;
#pragma unroll
    for (int dt = 0; dt < 4; ++dt)
#pragma unroll
      for (int r = 0; r < 8; ++r) o[dt][r] *= corr;

#if HAVE_TDM
    if (kv0 + 32 < NKV) __builtin_amdgcn_s_wait_tensorcnt(1);  // cur tile done
    else                __builtin_amdgcn_s_wait_tensorcnt(0);
#else
    __syncthreads();
#endif

    // ctx^T += V^T * P^T : four 16(d)x16(q) tiles, K = 32 kv
#pragma unroll
    for (int dt = 0; dt < 4; ++dt) {
      int d = dt * 16 + l15;
      v16h a;
#pragma unroll
      for (int i = 0; i < 8; ++i) {
        a[i]     = Vs[cur][(i +      hi * 8) * (HD + VPAD) + d];
        a[i + 8] = Vs[cur][(i + 16 + hi * 8) * (HD + VPAD) + d];
      }
      o[dt] = __builtin_amdgcn_wmma_f32_16x16x32_f16(false, a, false, pb, (short)0, o[dt], false, false);
    }
#if !HAVE_TDM
    __syncthreads();
#endif
  }

  float inv = 1.0f / lsum;
  h16* outp = CTX + ((size_t)bb * NQ + q0 + l15) * INNER + hh * HD;
#pragma unroll
  for (int dt = 0; dt < 4; ++dt)
#pragma unroll
    for (int r = 0; r < 8; ++r)
      outp[dt * 16 + r + hi * 8] = (h16)(o[dt][r] * inv);
}

// ---------------- launch ----------------
extern "C" void kernel_launch(void* const* d_in, const int* in_sizes, int n_in,
                              void* d_out, int out_size, void* d_ws, size_t ws_size,
                              hipStream_t stream) {
  const float* query     = (const float*)d_in[0];
  const float* key_value = (const float*)d_in[1];
  const float* w_q       = (const float*)d_in[2];
  const float* w_kv      = (const float*)d_in[3];
  const float* w_out     = (const float*)d_in[4];
  const float* b_out     = (const float*)d_in[5];
  float* out = (float*)d_out;

  char* ws = (char*)d_ws;
  h16* Xh   = (h16*)(ws + (size_t)0);          // query f16       [4096][1024]  8 MB
  h16* KVh  = (h16*)(ws + ((size_t)8  << 20)); // key_value f16   [4096][1024]  8 MB
  h16* wqT  = (h16*)(ws + ((size_t)16 << 20)); // w_q^T           [1024][1024]  2 MB
  h16* wkvT = (h16*)(ws + ((size_t)18 << 20)); // w_kv^T          [2048][1024]  4 MB
  h16* woT  = (h16*)(ws + ((size_t)22 << 20)); // w_out^T         [1024][1024]  2 MB
  h16* qh   = (h16*)(ws + ((size_t)24 << 20)); // q               [4096][1024]  8 MB
  h16* kvh  = (h16*)(ws + ((size_t)32 << 20)); // k|v             [4096][2048] 16 MB
  h16* ctxh = (h16*)(ws + ((size_t)48 << 20)); // ctx             [4096][1024]  8 MB

  f32_to_f16<<<dim3((MQ  * QDIM  + 255) / 256), dim3(256), 0, stream>>>(query,     Xh,  MQ * QDIM);
  f32_to_f16<<<dim3((MKV * KVDIM + 255) / 256), dim3(256), 0, stream>>>(key_value, KVh, MKV * KVDIM);
  transpose_f32_to_f16<<<dim3(INNER / 32, QDIM / 32),      dim3(32, 8), 0, stream>>>(w_q,   wqT,  QDIM,  INNER);
  transpose_f32_to_f16<<<dim3(2 * INNER / 32, KVDIM / 32), dim3(32, 8), 0, stream>>>(w_kv,  wkvT, KVDIM, 2 * INNER);
  transpose_f32_to_f16<<<dim3(QDIM / 32, INNER / 32),      dim3(32, 8), 0, stream>>>(w_out, woT,  INNER, QDIM);

  gemm_tt_h16<<<dim3(INNER / 64, MQ / 128), dim3(256), 0, stream>>>(Xh, wqT, qh, INNER, QDIM);
  gemm_tt_h16<<<dim3(2 * INNER / 64, MKV / 128), dim3(256), 0, stream>>>(KVh, wkvT, kvh, 2 * INNER, KVDIM);
  attn<<<dim3(NQ / 16, H, BATCH), dim3(32), 0, stream>>>(qh, kvh, ctxh);
  gemm_tt_f32_bias<<<dim3(QDIM / 64, MQ / 128), dim3(256), 0, stream>>>(ctxh, woT, b_out, out, QDIM, INNER);
}
